// MambaDecoder_48859547959720
// MI455X (gfx1250) — matrix operand order repaired and checked by reading, not compiled, not run on hardware
//
#include <hip/hip_runtime.h>
#include <hip/hip_bf16.h>

// ---------------------------------------------------------------------------
// Problem constants (from reference)
// ---------------------------------------------------------------------------
constexpr int BSZ     = 2;
constexpr int LSEQ    = 4096;
constexpr int MROW    = 2048;
constexpr int DMODEL  = 64;
constexpr int DINNER  = 128;   // EXPAND * D_MODEL
constexpr int DSTATE  = 64;
constexpr int DTRANK  = 4;
constexpr int DHID    = 32;
constexpr int NROWS   = BSZ * LSEQ;   // 8192 flattened (b, t) rows
constexpr int RPW     = 8;            // row tiles per wave in the GEMMs

typedef __attribute__((ext_vector_type(16))) _Float16 v16h;
typedef __attribute__((ext_vector_type(8)))  _Float16 v8h;
typedef __attribute__((ext_vector_type(8)))  float    v8f;

// ---------------------------------------------------------------------------
// WMMA fragment layouts (CDNA5 wave32, CDNA5 ISA):
//  A (16x32 f16): lane -> row (lane&15); per lane two contiguous 8-half
//                 chunks at K = base+hi and K = base+hi+16 (hi = lane>=16?8:0)
//  B (32x16 f16): lane -> col (lane&15). Element e -> K = (e<8?e:e+8) + (lane>=16?8:0)
//  C/D (16x16 f32): lane -> col (lane&15). Element r -> row = r+(lane>=16?8:0)
// ---------------------------------------------------------------------------
__device__ __forceinline__ int b_k(int lane, int e) {
    return (e < 8 ? e : e + 8) + (lane >= 16 ? 8 : 0);
}

__device__ __forceinline__ v8f wmma_f16(v16h a, v16h b, v8f c) {
    return __builtin_amdgcn_wmma_f32_16x16x32_f16(
        false, a, false, b, (short)0, c, false, false);
}

// Fast SiLU: v * v_rcp_f32(1 + exp(-v)) — avoids the IEEE divide expansion.
__device__ __forceinline__ float silu_f(float v) {
    return v * __builtin_amdgcn_rcpf(1.0f + __expf(-v));
}

// Async copy of one 16B chunk global -> LDS (ASYNCcnt-tracked, gfx1250).
__device__ __forceinline__ void async_copy16(unsigned ldsOff, const void* g) {
    asm volatile("global_load_async_to_lds_b128 %0, %1, off"
                 :: "v"(ldsOff), "v"((unsigned long long)(uintptr_t)g)
                 : "memory");
}

// Build an A fragment from an LDS tile with row stride `stride` halves.
__device__ __forceinline__ v16h a_frag_lds(const _Float16* tp, int lane,
                                           int kf, int stride) {
    const int lrow = lane & 15;
    const int hi   = (lane >= 16) ? 8 : 0;
    v8h lo  = *(const v8h*)&tp[lrow * stride + kf * 32 + hi];
    v8h hi8 = *(const v8h*)&tp[lrow * stride + kf * 32 + hi + 16];
    return __builtin_shufflevector(lo, hi8,
        0,1,2,3,4,5,6,7,8,9,10,11,12,13,14,15);
}

// ---------------------------------------------------------------------------
// P0: pack a weight matrix W (rows x K, row-major f32) into f16 B-fragment
// layout: out[((ct*nkf + kf)*32 + lane)*16 + e] = W[rowOff + ct*16 + (lane&15)]
//                                                  [kf*32 + b_k(lane,e)]
// ---------------------------------------------------------------------------
__global__ void k_pack(const float* __restrict__ W, _Float16* __restrict__ out,
                       int K, int nkf, int rowOff)
{
    const int lane = threadIdx.x & 31;
    const int kf   = blockIdx.x % nkf;
    const int ct   = blockIdx.x / nkf;
    const int n    = rowOff + ct * 16 + (lane & 15);
    v16h v;
#pragma unroll
    for (int e = 0; e < 16; ++e)
        v[e] = (_Float16)W[n * K + kf * 32 + b_k(lane, e)];
    *(v16h*)&out[((size_t)blockIdx.x * 32 + lane) * 16] = v;
}

// ---------------------------------------------------------------------------
// P1: u = decoded_reads @ exp_w^T + exp_b  (K=4) -> f16 buffer (NROWS,64)
// ---------------------------------------------------------------------------
__global__ void k_u(const float* __restrict__ dreads, const float* __restrict__ exp_w,
                    const float* __restrict__ exp_b, _Float16* __restrict__ uh)
{
    const int idx = blockIdx.x * blockDim.x + threadIdx.x;   // NROWS*64
    const int row = idx >> 6, k = idx & 63;
    float u = exp_b[k];
#pragma unroll
    for (int j = 0; j < 4; ++j)
        u += dreads[row * 4 + j] * exp_w[k * 4 + j];
    uh[idx] = (_Float16)u;
}

// ---------------------------------------------------------------------------
// K1: xz = u @ in_proj_w^T (M=8192, K=64, N=256), WMMA f16.
// A-tiles (16x64 f16, 2 KB) staged to LDS via async-to-LDS, double-buffered;
// B fragments loaded once per wave and reused across RPW row tiles.
// Column tile is entirely in the x half (ct<8) or z half (ct>=8) -> the
// epilogue select is a UNIFORM branch (no exec-mask divergence).
// grid (NROWS/(16*RPW), 16), block 32.
// ---------------------------------------------------------------------------
__global__ void k_inproj(const _Float16* __restrict__ uh,
                         const _Float16* __restrict__ wpack,
                         const float* __restrict__ conv_w,
                         const float* __restrict__ conv_b,
                         float* __restrict__ xbuf, float* __restrict__ zbuf,
                         _Float16* __restrict__ xh)
{
    __shared__ _Float16 tileA[2][16 * DMODEL];
    const int lane  = threadIdx.x & 31;
    const int ct    = blockIdx.y;
    const int rbase = blockIdx.x * RPW;
    const bool isx  = (ct < 8);            // uniform: whole tile in x half

    const v16h b0 = *(const v16h*)&wpack[((size_t)(ct * 2 + 0) * 32 + lane) * 16];
    const v16h b1 = *(const v16h*)&wpack[((size_t)(ct * 2 + 1) * 32 + lane) * 16];

    const int  ncol = ct * 16 + (lane & 15);
    const float cw  = isx ? conv_w[ncol] : 0.0f;
    const float cb  = isx ? conv_b[ncol] : 0.0f;
    const int  zcol = ncol - DINNER;

    const unsigned lds0 = (unsigned)(uintptr_t)&tileA[0][0];
    const unsigned lds1 = (unsigned)(uintptr_t)&tileA[1][0];

    // 2 KB tile = 128 x 16B chunks -> 4 per lane
    auto issue = [&](unsigned ldsBase, int tileIdx) {
        const char* g = (const char*)uh + (size_t)tileIdx * 16 * DMODEL * 2;
#pragma unroll
        for (int i = 0; i < 4; ++i)
            async_copy16(ldsBase + (i * 32 + lane) * 16, g + (i * 32 + lane) * 16);
    };

    issue(lds0, rbase);
    for (int rt = 0; rt < RPW; ++rt) {
        if (rt + 1 < RPW) {
            issue(((rt + 1) & 1) ? lds1 : lds0, rbase + rt + 1);
            asm volatile("s_wait_asynccnt 0x4" ::: "memory"); // current tile done
        } else {
            asm volatile("s_wait_asynccnt 0x0" ::: "memory");
        }
        const _Float16* tp = &tileA[rt & 1][0];
        v8f acc = {};
        acc = wmma_f16(a_frag_lds(tp, lane, 0, DMODEL), b0, acc);
        acc = wmma_f16(a_frag_lds(tp, lane, 1, DMODEL), b1, acc);

        const int r0 = (rbase + rt) * 16;
        if (isx) {
#pragma unroll
            for (int r = 0; r < 8; ++r) {
                const int row = r0 + r + (lane >= 16 ? 8 : 0);
                const float s = silu_f(acc[r] * cw + cb);
                xbuf[row * DINNER + ncol] = s;
                xh  [row * DINNER + ncol] = (_Float16)s;
            }
        } else {
#pragma unroll
            for (int r = 0; r < 8; ++r) {
                const int row = r0 + r + (lane >= 16 ? 8 : 0);
                zbuf[row * DINNER + zcol] = acc[r];
            }
        }
    }
}

// ---------------------------------------------------------------------------
// K2: B/C columns of x_dbl = x @ x_proj_w^T (M=8192, K=128, N=128), WMMA f16.
// Same async double-buffered LDS staging; 4 B fragments held in registers.
// Column tile is entirely B (ct<4) or C (ct>=4) -> uniform epilogue branch.
// grid (NROWS/(16*RPW), 8), block 32.
// ---------------------------------------------------------------------------
__global__ void k_xproj(const _Float16* __restrict__ xh,
                        const _Float16* __restrict__ wpack,
                        float* __restrict__ Bbuf, float* __restrict__ Cbuf)
{
    __shared__ _Float16 tileA[2][16 * DINNER];
    const int lane  = threadIdx.x & 31;
    const int ct    = blockIdx.y;
    const int rbase = blockIdx.x * RPW;

    v16h bf[4];
#pragma unroll
    for (int kf = 0; kf < 4; ++kf)
        bf[kf] = *(const v16h*)&wpack[((size_t)(ct * 4 + kf) * 32 + lane) * 16];

    const unsigned lds0 = (unsigned)(uintptr_t)&tileA[0][0];
    const unsigned lds1 = (unsigned)(uintptr_t)&tileA[1][0];

    // 4 KB tile = 256 x 16B chunks -> 8 per lane
    auto issue = [&](unsigned ldsBase, int tileIdx) {
        const char* g = (const char*)xh + (size_t)tileIdx * 16 * DINNER * 2;
#pragma unroll
        for (int i = 0; i < 8; ++i)
            async_copy16(ldsBase + (i * 32 + lane) * 16, g + (i * 32 + lane) * 16);
    };

    // uniform select of destination half (whole 16-col tile in one half)
    float* __restrict__ dst = (ct < 4) ? Bbuf : Cbuf;
    const int n = ((ct < 4) ? ct : ct - 4) * 16 + (lane & 15);

    issue(lds0, rbase);
    for (int rt = 0; rt < RPW; ++rt) {
        if (rt + 1 < RPW) {
            issue(((rt + 1) & 1) ? lds1 : lds0, rbase + rt + 1);
            asm volatile("s_wait_asynccnt 0x8" ::: "memory");
        } else {
            asm volatile("s_wait_asynccnt 0x0" ::: "memory");
        }
        const _Float16* tp = &tileA[rt & 1][0];
        v8f acc = {};
#pragma unroll
        for (int kf = 0; kf < 4; ++kf)
            acc = wmma_f16(a_frag_lds(tp, lane, kf, DINNER), bf[kf], acc);

        const int r0 = (rbase + rt) * 16;
#pragma unroll
        for (int r = 0; r < 8; ++r) {
            const int row = r0 + r + (lane >= 16 ? 8 : 0);
            dst[row * DSTATE + n] = acc[r];
        }
    }
}

// ---------------------------------------------------------------------------
// K2b: dt = x @ x_proj_w[0:4]^T ; delta = softplus(dt @ dt_proj_w^T + b).
// ---------------------------------------------------------------------------
__global__ void k_delta(const float* __restrict__ xbuf,
                        const float* __restrict__ xpw,
                        const float* __restrict__ dtw,
                        const float* __restrict__ dtb,
                        float* __restrict__ dbuf)
{
    __shared__ float dt[DTRANK];
    const int row = blockIdx.x;
    const int tid = threadIdx.x;

    if (tid < DTRANK) {
        float s = 0.0f;
        for (int k = 0; k < DINNER; ++k)
            s += xbuf[row * DINNER + k] * xpw[tid * DINNER + k];
        dt[tid] = s;
    }
    __syncthreads();

    float s = dtb[tid];
#pragma unroll
    for (int j = 0; j < DTRANK; ++j) s += dt[j] * dtw[tid * DTRANK + j];
    dbuf[row * DINNER + tid] = (s > 20.0f) ? s : __logf(1.0f + __expf(s));
}

// ---------------------------------------------------------------------------
// K3: selective scan. One wave32 per (channel d, batch b); lane owns state
// elements s and s+32; dA computed on the fly; y reduced in-register to
// ybar / ylast (pooling commutes with the linear out-projection).
// ---------------------------------------------------------------------------
__global__ void k_scan(const float* __restrict__ xbuf,
                       const float* __restrict__ zbuf,
                       const float* __restrict__ dbuf,
                       const float* __restrict__ Bbuf,
                       const float* __restrict__ Cbuf,
                       const float* __restrict__ A_log,
                       const float* __restrict__ Dvec,
                       float* __restrict__ ybar,
                       float* __restrict__ ylast)
{
    const int d    = blockIdx.x;
    const int b    = blockIdx.y;
    const int lane = threadIdx.x;

    const float A0 = -__expf(A_log[d * DSTATE + lane]);
    const float A1 = -__expf(A_log[d * DSTATE + lane + 32]);
    const float Dd = Dvec[d];

    float h0 = 0.0f, h1 = 0.0f, ysum = 0.0f, ylv = 0.0f;
    const int base = b * LSEQ;

    for (int t = 0; t < LSEQ; ++t) {
        const int r = base + t;
        const float delta = dbuf[r * DINNER + d];
        const float xv    = xbuf[r * DINNER + d];
        const float zv    = zbuf[r * DINNER + d];
        const float Bv0   = Bbuf[r * DSTATE + lane];
        const float Bv1   = Bbuf[r * DSTATE + lane + 32];
        const float Cv0   = Cbuf[r * DSTATE + lane];
        const float Cv1   = Cbuf[r * DSTATE + lane + 32];

        const float dbu = delta * xv;
        h0 = __expf(delta * A0) * h0 + dbu * Bv0;
        h1 = __expf(delta * A1) * h1 + dbu * Bv1;

        float p = h0 * Cv0 + h1 * Cv1;
#pragma unroll
        for (int off = 16; off > 0; off >>= 1)
            p += __shfl_xor(p, off, 32);

        float y = p + xv * Dd;
        y *= silu_f(zv);
        ysum += y;
        if (t == LSEQ - 1) ylv = y;
    }

    if (lane == 0) {
        ybar [b * DINNER + d] = ysum * (1.0f / (float)LSEQ);
        ylast[b * DINNER + d] = ylv;
    }
}

// ---------------------------------------------------------------------------
// K4: pooled = [ybar @ out_w^T , ylast @ out_w^T]; queries = pooled @ q_w^T+b
// ---------------------------------------------------------------------------
__global__ void k_head(const float* __restrict__ ybar,
                       const float* __restrict__ ylast,
                       const float* __restrict__ out_w,
                       const float* __restrict__ q_w,
                       const float* __restrict__ q_b,
                       float* __restrict__ queries)
{
    __shared__ float pooled[BSZ][2 * DMODEL];
    const int tid = threadIdx.x;           // 128 threads
    const int n   = tid >> 6;
    const int c   = tid & 63;

    float sm = 0.0f, sl = 0.0f;
    for (int k = 0; k < DINNER; ++k) {
        const float w = out_w[c * DINNER + k];
        sm += ybar [n * DINNER + k] * w;
        sl += ylast[n * DINNER + k] * w;
    }
    pooled[n][c]          = sm;
    pooled[n][DMODEL + c] = sl;
    __syncthreads();

    if (tid < BSZ * DHID) {
        const int nn = tid >> 5, h = tid & 31;
        float s = q_b[h];
        for (int k = 0; k < 2 * DMODEL; ++k)
            s += pooled[nn][k] * q_w[h * 2 * DMODEL + k];
        queries[nn * DHID + h] = s;
    }
}

// ---------------------------------------------------------------------------
// K5: keys = encoded_reads @ k_w^T + k_b; out[n][m] = queries[n] . keys[m]
// ---------------------------------------------------------------------------
__global__ void k_out(const float* __restrict__ enc,
                      const float* __restrict__ k_w,
                      const float* __restrict__ k_b,
                      const float* __restrict__ queries,
                      float* __restrict__ out)
{
    const int m = blockIdx.x * blockDim.x + threadIdx.x;
    if (m >= MROW) return;
    float o0 = 0.0f, o1 = 0.0f;
    for (int h = 0; h < DHID; ++h) {
        float s = k_b[h];
        for (int k = 0; k < 2 * DMODEL; ++k)
            s += enc[m * 2 * DMODEL + k] * k_w[h * 2 * DMODEL + k];
        o0 += queries[h] * s;
        o1 += queries[DHID + h] * s;
    }
    out[m]        = o0;
    out[MROW + m] = o1;
}

// ---------------------------------------------------------------------------
// Launcher
// ---------------------------------------------------------------------------
extern "C" void kernel_launch(void* const* d_in, const int* in_sizes, int n_in,
                              void* d_out, int out_size, void* d_ws, size_t ws_size,
                              hipStream_t stream) {
    const float* enc    = (const float*)d_in[0];
    const float* dreads = (const float*)d_in[1];
    const float* exp_w  = (const float*)d_in[2];
    const float* exp_b  = (const float*)d_in[3];
    const float* in_w   = (const float*)d_in[4];
    const float* conv_w = (const float*)d_in[5];
    const float* conv_b = (const float*)d_in[6];
    const float* xpw    = (const float*)d_in[7];
    const float* dtw    = (const float*)d_in[8];
    const float* dtb    = (const float*)d_in[9];
    const float* A_log  = (const float*)d_in[10];
    const float* Dvec   = (const float*)d_in[11];
    const float* out_w  = (const float*)d_in[12];
    const float* q_w    = (const float*)d_in[13];
    const float* q_b    = (const float*)d_in[14];
    const float* k_w    = (const float*)d_in[15];
    const float* k_b    = (const float*)d_in[16];
    float* out = (float*)d_out;

    // Workspace carve-up: f32 region then f16 region (all 32B-aligned).
    float* xbuf  = (float*)d_ws;                              // NROWS*128
    float* zbuf  = xbuf  + (size_t)NROWS * DINNER;            // NROWS*128
    float* dbuf  = zbuf  + (size_t)NROWS * DINNER;            // NROWS*128
    float* Bbuf  = dbuf  + (size_t)NROWS * DINNER;            // NROWS*64
    float* Cbuf  = Bbuf  + (size_t)NROWS * DSTATE;            // NROWS*64
    float* ybar  = Cbuf  + (size_t)NROWS * DSTATE;            // 2*128
    float* ylast = ybar  + BSZ * DINNER;                      // 2*128
    float* qbuf  = ylast + BSZ * DINNER;                      // 64
    _Float16* uh      = (_Float16*)(qbuf + 64);               // NROWS*64
    _Float16* xh      = uh + (size_t)NROWS * DMODEL;          // NROWS*128
    _Float16* wpack_i = xh + (size_t)NROWS * DINNER;          // 16*2*32*16
    _Float16* wpack_x = wpack_i + 16 * 2 * 32 * 16;           // 8*4*32*16

    // P0: pack weights into f16 WMMA B-fragment layout
    k_pack<<<16 * 2, 32, 0, stream>>>(in_w, wpack_i, DMODEL, 2, 0);
    k_pack<<< 8 * 4, 32, 0, stream>>>(xpw, wpack_x, DINNER, 4, DTRANK);

    // P1: u projection (K=4) -> f16
    k_u<<<NROWS * DMODEL / 256, 256, 0, stream>>>(dreads, exp_w, exp_b, uh);

    // K1: in_proj GEMM (WMMA, async LDS staging) + conv/SiLU epilogue
    k_inproj<<<dim3(NROWS / (16 * RPW), (2 * DINNER) / 16), 32, 0, stream>>>(
        uh, wpack_i, conv_w, conv_b, xbuf, zbuf, xh);

    // K2: x_proj GEMM for B/C (WMMA, async LDS staging)
    k_xproj<<<dim3(NROWS / (16 * RPW), (2 * DSTATE) / 16), 32, 0, stream>>>(
        xh, wpack_x, Bbuf, Cbuf);

    // K2b: dt + softplus(delta)
    k_delta<<<NROWS, DINNER, 0, stream>>>(xbuf, xpw, dtw, dtb, dbuf);

    // K3: selective scan with in-register pooling
    k_scan<<<dim3(DINNER, BSZ), 32, 0, stream>>>(
        xbuf, zbuf, dbuf, Bbuf, Cbuf, A_log, Dvec, ybar, ylast);

    // K4: pooled -> queries
    k_head<<<1, 128, 0, stream>>>(ybar, ylast, out_w, q_w, q_b, qbuf);

    // K5: keys + final score matrix
    k_out<<<MROW / 256, 256, 0, stream>>>(enc, k_w, k_b, qbuf, out);
}
